// RecurrentMoE_86268713107990
// MI455X (gfx1250) — 8-bit matrix operand order repaired and checked by reading, compiled, not run on hardware
//
#include <hip/hip_runtime.h>

#define BB 256
#define TT 32
#define DD 1024
#define HH 1024
#define EE 8
#define GG 4096

#define SCALE_W   64.0f
#define SCALE_A   16.0f
#define OUT_SCALE (1.0f / 1024.0f)

typedef unsigned char fp8_t;
typedef int   v2i __attribute__((ext_vector_type(2)));
typedef int   v4i __attribute__((ext_vector_type(4)));
typedef int   v8i __attribute__((ext_vector_type(8)));
typedef float v8f __attribute__((ext_vector_type(8)));

__device__ __forceinline__ unsigned int f32_to_e4m3(float x) {
  unsigned int u = __float_as_uint(x);
  unsigned int s = (u >> 24) & 0x80u;
  unsigned int a = u & 0x7fffffffu;
  if (a >= 0x7f800000u) return s | 0x7fu;
  int exp = (int)(a >> 23) - 127;
  unsigned int man = a & 0x7fffffu;
  if (exp >= -6) {
    unsigned int keep = man >> 20;
    unsigned int rest = man & 0xfffffu;
    keep += (rest > 0x80000u) || ((rest == 0x80000u) && (keep & 1u));
    int E = exp + 7;
    if (keep == 8u) { keep = 0u; ++E; }
    if (E > 15 || (E == 15 && keep > 6u)) return s | 0x7eu;
    return s | ((unsigned int)E << 3) | keep;
  }
  if (exp < -10) return s;
  unsigned int full = 0x800000u | man;
  int rs = 20 + (-6 - exp);
  unsigned int keep = full >> rs;
  unsigned int rem = full & ((1u << rs) - 1u);
  unsigned int halfv = 1u << (rs - 1);
  keep += (rem > halfv) || ((rem == halfv) && (keep & 1u));
  if (keep >= 8u) return s | (1u << 3);
  return s | keep;
}

__global__ void cvt_f32_fp8_kernel(const float* __restrict__ s,
                                   unsigned int* __restrict__ d, float scale,
                                   long n4) {
  long i = (long)blockIdx.x * blockDim.x + threadIdx.x;
  long st = (long)gridDim.x * blockDim.x;
  for (; i < n4; i += st) {
    float4 v = ((const float4*)s)[i];
    d[i] = f32_to_e4m3(v.x * scale) | (f32_to_e4m3(v.y * scale) << 8) |
           (f32_to_e4m3(v.z * scale) << 16) | (f32_to_e4m3(v.w * scale) << 24);
  }
}

__global__ void zero16_kernel(uint4* __restrict__ p, long n16) {
  long i = (long)blockIdx.x * blockDim.x + threadIdx.x;
  long st = (long)gridDim.x * blockDim.x;
  for (; i < n16; i += st) p[i] = make_uint4(0u, 0u, 0u, 0u);
}

__global__ __launch_bounds__(256) void gemm_wmma_fp8_kernel(
    const fp8_t* __restrict__ A1, long a1RowStride, long a1EStride,
    const fp8_t* __restrict__ W1, long w1EStride,
    const fp8_t* __restrict__ A2, long a2EStride,
    const fp8_t* __restrict__ W2, long w2EStride,
    const float* __restrict__ bias, long biasEStride, float outScale,
    float* __restrict__ C, long cEStride, int ldc) {
  const int e    = blockIdx.y;
  const int wave = threadIdx.x >> 5;
  const int lane = threadIdx.x & 31;
  const int nsel = lane & 15;
  const int half = lane >> 4;
  const int n0   = blockIdx.x * 64;
  const int m0   = wave * 32;

  v8f acc[2][4];
#pragma unroll
  for (int i = 0; i < 2; ++i)
#pragma unroll
    for (int j = 0; j < 4; ++j)
      acc[i][j] = v8f{0.f, 0.f, 0.f, 0.f, 0.f, 0.f, 0.f, 0.f};

#pragma unroll 1
  for (int ph = 0; ph < 2; ++ph) {
    const fp8_t* Ab; const fp8_t* Wb; long aStr;
    if (ph == 0) {
      Ab = A1 + (long)e * a1EStride; Wb = W1 + (long)e * w1EStride;
      aStr = a1RowStride;
    } else {
      if (A2 == nullptr) break;
      Ab = A2 + (long)e * a2EStride; Wb = W2 + (long)e * w2EStride;
      aStr = HH;
    }
#pragma unroll 1
    for (int k = 0; k < 1024; k += 64) {
      v8i a[2];
#pragma unroll
      for (int mt = 0; mt < 2; ++mt) {
        const fp8_t* ar =
            Ab + (long)(m0 + mt * 16 + nsel) * aStr + k + 16 * half;
        v2i q0 = *(const v2i*)(ar);
        v2i q1 = *(const v2i*)(ar + 8);
        v2i q2 = *(const v2i*)(ar + 32);
        v2i q3 = *(const v2i*)(ar + 40);
        v4i lo = __builtin_shufflevector(q0, q1, 0, 1, 2, 3);
        v4i hi = __builtin_shufflevector(q2, q3, 0, 1, 2, 3);
        a[mt] = __builtin_shufflevector(lo, hi, 0, 1, 2, 3, 4, 5, 6, 7);
      }
#pragma unroll
      for (int nt = 0; nt < 4; ++nt) {
        const fp8_t* br =
            Wb + (long)(n0 + nt * 16 + nsel) * 1024 + k + 16 * half;
        v4i b0 = *(const v4i*)(br);
        v4i b1 = *(const v4i*)(br + 32);
        v8i b = __builtin_shufflevector(b0, b1, 0, 1, 2, 3, 4, 5, 6, 7);
#pragma unroll
        for (int mt = 0; mt < 2; ++mt)
          acc[mt][nt] = __builtin_amdgcn_wmma_f32_16x16x64_fp8_fp8(
              a[mt], b, (short)0, acc[mt][nt], false, false);
      }
    }
  }

  float* Cb = C + (long)e * cEStride;
#pragma unroll
  for (int mt = 0; mt < 2; ++mt)
#pragma unroll
    for (int nt = 0; nt < 4; ++nt) {
      const int col = n0 + nt * 16 + nsel;
      const float bv = bias ? bias[(long)e * biasEStride + col] : 0.f;
#pragma unroll
      for (int r = 0; r < 8; ++r) {
        const int row = m0 + mt * 16 + r + 8 * half;
        Cb[(long)row * ldc + col] = acc[mt][nt][r] * outScale + bv;
      }
    }
}


__device__ __forceinline__ float sigmoidf_(float x) {
  return 1.f / (1.f + __expf(-x));
}

__global__ __launch_bounds__(256) void lstm_pointwise_kernel(
    const float* __restrict__ G, long gEStride,
    const float* __restrict__ bih, const float* __restrict__ bhh, long bEStride,
    const float* __restrict__ cOld, long cInEStride,
    float* __restrict__ cNew, float* __restrict__ hNew, long outEStride,
    fp8_t* __restrict__ hQ, long hqEStride) {
  const int e = blockIdx.y;
  const int idx = blockIdx.x * blockDim.x + threadIdx.x;
  const int hh = idx & (HH - 1);
  const float* g = G + (long)e * gEStride + (long)(idx >> 10) * GG;
  const float* bi = bih + (long)e * bEStride;
  const float* bh = bhh + (long)e * bEStride;
  float iv = sigmoidf_(g[hh]          + bi[hh]          + bh[hh]);
  float fv = sigmoidf_(g[HH + hh]     + bi[HH + hh]     + bh[HH + hh]);
  float gv = tanhf    (g[2 * HH + hh] + bi[2 * HH + hh] + bh[2 * HH + hh]);
  float ov = sigmoidf_(g[3 * HH + hh] + bi[3 * HH + hh] + bh[3 * HH + hh]);
  float c = fv * cOld[(long)e * cInEStride + idx] + iv * gv;
  float h = ov * tanhf(c);
  cNew[(long)e * outEStride + idx] = c;
  hNew[(long)e * outEStride + idx] = h;
  hQ[(long)e * hqEStride + idx] = (fp8_t)f32_to_e4m3(h * SCALE_A);
}


__global__ __launch_bounds__(64) void gate_topk_kernel(
    const float* __restrict__ c2, const float* __restrict__ Wg,
    const float* __restrict__ bg, float* __restrict__ wOut,
    int* __restrict__ selOut) {
  const int b = blockIdx.x, t = threadIdx.x;
  float p[EE];
#pragma unroll
  for (int e = 0; e < EE; ++e) p[e] = 0.f;
  for (int h = t; h < HH; h += 64) {
    float cv = c2[(long)b * HH + h];
#pragma unroll
    for (int e = 0; e < EE; ++e) p[e] += cv * Wg[(long)e * HH + h];
  }
  __shared__ float red[EE][64];
#pragma unroll
  for (int e = 0; e < EE; ++e) red[e][t] = p[e];
  __syncthreads();
  if (t == 0) {
    float pr[EE];
    float m = -1e30f;
    for (int e = 0; e < EE; ++e) {
      float s = 0.f;
      for (int j = 0; j < 64; ++j) s += red[e][j];
      pr[e] = s + bg[e];
      m = pr[e] > m ? pr[e] : m;
    }
    float den = 0.f;
    for (int e = 0; e < EE; ++e) { pr[e] = __expf(pr[e] - m); den += pr[e]; }
    for (int e = 0; e < EE; ++e) pr[e] /= den;
    int i0 = 0;
    for (int e = 1; e < EE; ++e) if (pr[e] > pr[i0]) i0 = e;
    int i1 = (i0 == 0) ? 1 : 0;
    for (int e = 0; e < EE; ++e)
      if (e != i0 && pr[e] > pr[i1]) i1 = e;
    wOut[b * 2] = pr[i0]; wOut[b * 2 + 1] = pr[i1];
    selOut[b * 2] = i0;   selOut[b * 2 + 1] = i1;
  }
}


__global__ __launch_bounds__(256) void expert_state_update_kernel(
    const int* __restrict__ sel, const float* __restrict__ h1n,
    const float* __restrict__ c1n, const float* __restrict__ h2n,
    const float* __restrict__ c2n, const fp8_t* __restrict__ h1nq,
    const fp8_t* __restrict__ h2nq, float* __restrict__ h1,
    float* __restrict__ c1, float* __restrict__ h2, float* __restrict__ c2,
    fp8_t* __restrict__ h1q, fp8_t* __restrict__ h2q) {
  const int e = blockIdx.y;
  const int idx = blockIdx.x * blockDim.x + threadIdx.x;
  const int b = idx >> 10;
  if (sel[b * 2] == e || sel[b * 2 + 1] == e) {
    const long o = (long)e * ((long)BB * HH) + idx;
    h1[o] = h1n[o]; c1[o] = c1n[o]; h2[o] = h2n[o]; c2[o] = c2n[o];
    h1q[o] = h1nq[o]; h2q[o] = h2nq[o];
  }
}

__global__ __launch_bounds__(256) void combine_kernel(
    const float* __restrict__ y, const float* __restrict__ w,
    const int* __restrict__ sel, float* __restrict__ out, int t) {
  const int idx = blockIdx.x * blockDim.x + threadIdx.x;
  const int b = idx >> 10, d = idx & (DD - 1);
  const long sz = (long)BB * DD;
  const int s0 = sel[b * 2], s1 = sel[b * 2 + 1];
  float v = w[b * 2] * y[(long)s0 * sz + idx] +
            w[b * 2 + 1] * y[(long)s1 * sz + idx];
  out[((long)b * TT + t) * DD + d] = v;
}


extern "C" void kernel_launch(void* const* d_in, const int* in_sizes, int n_in,
                              void* d_out, int out_size, void* d_ws,
                              size_t ws_size, hipStream_t stream) {
  const float* x      = (const float*)d_in[0];
  const float* dWih1  = (const float*)d_in[1];
  const float* dWhh1  = (const float*)d_in[2];
  const float* dbih1  = (const float*)d_in[3];
  const float* dbhh1  = (const float*)d_in[4];
  const float* dWih2  = (const float*)d_in[5];
  const float* dWhh2  = (const float*)d_in[6];
  const float* dbih2  = (const float*)d_in[7];
  const float* dbhh2  = (const float*)d_in[8];
  const float* eWih1  = (const float*)d_in[11];
  const float* eWhh1  = (const float*)d_in[12];
  const float* ebih1  = (const float*)d_in[13];
  const float* ebhh1  = (const float*)d_in[14];
  const float* eWih2  = (const float*)d_in[15];
  const float* eWhh2  = (const float*)d_in[16];
  const float* ebih2  = (const float*)d_in[17];
  const float* ebhh2  = (const float*)d_in[18];
  const float* eWout  = (const float*)d_in[19];
  const float* ebout  = (const float*)d_in[20];
  const float* gWg    = (const float*)d_in[21];
  const float* gbg    = (const float*)d_in[22];
  float* out = (float*)d_out;

  const long SZ_BH = (long)BB * HH;
  const long SZ_G  = (long)BB * GG;
  const long SZ_W  = (long)GG * HH;
  const long SZ_WO = (long)DD * HH;
  const long SZ_X  = (long)BB * TT * DD;

  char* ws = (char*)d_ws;
  size_t off = 0;
  auto take = [&](size_t bytes) -> char* {
    char* p = ws + off;
    off += (bytes + 255) & ~(size_t)255;
    return p;
  };

  fp8_t* xq      = (fp8_t*)take(SZ_X);
  fp8_t* dWih1q  = (fp8_t*)take(SZ_W);
  fp8_t* dWhh1q  = (fp8_t*)take(SZ_W);
  fp8_t* dWih2q  = (fp8_t*)take(SZ_W);
  fp8_t* dWhh2q  = (fp8_t*)take(SZ_W);
  fp8_t* eWih1q  = (fp8_t*)take(EE * SZ_W);
  fp8_t* eWhh1q  = (fp8_t*)take(EE * SZ_W);
  fp8_t* eWih2q  = (fp8_t*)take(EE * SZ_W);
  fp8_t* eWhh2q  = (fp8_t*)take(EE * SZ_W);
  fp8_t* eWoutq  = (fp8_t*)take(EE * SZ_WO);
  const long stateBytes = SZ_BH * 162;
  char* sb = take(stateBytes);
  float* dh1 = (float*)sb;
  float* dc1 = dh1 + SZ_BH;
  float* dh2 = dc1 + SZ_BH;
  float* dc2 = dh2 + SZ_BH;
  float* eh1 = dc2 + SZ_BH;
  float* ec1 = eh1 + EE * SZ_BH;
  float* eh2 = ec1 + EE * SZ_BH;
  float* ec2 = eh2 + EE * SZ_BH;
  fp8_t* dh1q = (fp8_t*)(ec2 + EE * SZ_BH);
  fp8_t* dh2q = dh1q + SZ_BH;
  fp8_t* eh1q = dh2q + SZ_BH;
  fp8_t* eh2q = eh1q + EE * SZ_BH;
  float* h1n  = (float*)take(EE * SZ_BH * 4);
  float* c1n  = (float*)take(EE * SZ_BH * 4);
  float* h2n  = (float*)take(EE * SZ_BH * 4);
  float* c2n  = (float*)take(EE * SZ_BH * 4);
  fp8_t* h1nq = (fp8_t*)take(EE * SZ_BH);
  fp8_t* h2nq = (fp8_t*)take(EE * SZ_BH);
  float* Gd   = (float*)take(SZ_G * 4);
  float* Ge   = (float*)take(EE * SZ_G * 4);
  float* ybuf = (float*)take(EE * SZ_BH * 4);
  float* wbuf = (float*)take(BB * 2 * 4);
  int*   selb = (int*)take(BB * 2 * 4);
  if (off > ws_size) return;

  auto cvt = [&](const float* s, fp8_t* d, float scale, long n) {
    long nb = (n / 4 + 255) / 256; if (nb > 4096) nb = 4096;
    cvt_f32_fp8_kernel<<<(int)nb, 256, 0, stream>>>(s, (unsigned int*)d,
                                                    scale, n / 4);
  };
  cvt(x, xq, SCALE_A, SZ_X);
  cvt(dWih1, dWih1q, SCALE_W, SZ_W); cvt(dWhh1, dWhh1q, SCALE_W, SZ_W);
  cvt(dWih2, dWih2q, SCALE_W, SZ_W); cvt(dWhh2, dWhh2q, SCALE_W, SZ_W);
  cvt(eWih1, eWih1q, SCALE_W, EE * SZ_W); cvt(eWhh1, eWhh1q, SCALE_W, EE * SZ_W);
  cvt(eWih2, eWih2q, SCALE_W, EE * SZ_W); cvt(eWhh2, eWhh2q, SCALE_W, EE * SZ_W);
  cvt(eWout, eWoutq, SCALE_W, EE * SZ_WO);
  zero16_kernel<<<4096, 256, 0, stream>>>((uint4*)sb, stateBytes / 16);

  const dim3 blk(256);
  const dim3 gGemmD(GG / 64, 1), gGemmE(GG / 64, EE), gGemmO(DD / 64, EE);
  const dim3 gPwD(SZ_BH / 256, 1), gPwE(SZ_BH / 256, EE);

  for (int t = 0; t < TT; ++t) {
    const fp8_t* xt = xq + (long)t * DD;
    gemm_wmma_fp8_kernel<<<gGemmD, blk, 0, stream>>>(
        xt, (long)TT * DD, 0, dWih1q, 0, dh1q, 0, dWhh1q, 0, nullptr, 0,
        OUT_SCALE, Gd, 0, GG);
    lstm_pointwise_kernel<<<gPwD, blk, 0, stream>>>(
        Gd, 0, dbih1, dbhh1, 0, dc1, 0, dc1, dh1, 0, dh1q, 0);
    gemm_wmma_fp8_kernel<<<gGemmD, blk, 0, stream>>>(
        dh1q, (long)HH, 0, dWih2q, 0, dh2q, 0, dWhh2q, 0, nullptr, 0,
        OUT_SCALE, Gd, 0, GG);
    lstm_pointwise_kernel<<<gPwD, blk, 0, stream>>>(
        Gd, 0, dbih2, dbhh2, 0, dc2, 0, dc2, dh2, 0, dh2q, 0);
    gate_topk_kernel<<<BB, 64, 0, stream>>>(dc2, gWg, gbg, wbuf, selb);
    gemm_wmma_fp8_kernel<<<gGemmE, blk, 0, stream>>>(
        xt, (long)TT * DD, 0, eWih1q, SZ_W, eh1q, SZ_BH, eWhh1q, SZ_W,
        nullptr, 0, OUT_SCALE, Ge, SZ_G, GG);
    lstm_pointwise_kernel<<<gPwE, blk, 0, stream>>>(
        Ge, SZ_G, ebih1, ebhh1, GG, ec1, SZ_BH, c1n, h1n, SZ_BH, h1nq, SZ_BH);
    gemm_wmma_fp8_kernel<<<gGemmE, blk, 0, stream>>>(
        h1nq, (long)HH, SZ_BH, eWih2q, SZ_W, eh2q, SZ_BH, eWhh2q, SZ_W,
        nullptr, 0, OUT_SCALE, Ge, SZ_G, GG);
    lstm_pointwise_kernel<<<gPwE, blk, 0, stream>>>(
        Ge, SZ_G, ebih2, ebhh2, GG, ec2, SZ_BH, c2n, h2n, SZ_BH, h2nq, SZ_BH);
    gemm_wmma_fp8_kernel<<<gGemmO, blk, 0, stream>>>(
        h2nq, (long)HH, SZ_BH, eWoutq, SZ_WO, nullptr, 0, nullptr, 0,
        ebout, DD, OUT_SCALE, ybuf, SZ_BH, DD);
    expert_state_update_kernel<<<gPwE, blk, 0, stream>>>(
        selb, h1n, c1n, h2n, c2n, h1nq, h2nq, eh1, ec1, eh2, ec2, eh1q, eh2q);
    combine_kernel<<<SZ_BH / 256, blk, 0, stream>>>(ybuf, wbuf, selb, out, t);
  }
}
